// Imputation_77524159693049
// MI455X (gfx1250) — matrix operand order repaired and checked by reading, not compiled, not run on hardware
//
#include <hip/hip_runtime.h>

typedef __attribute__((ext_vector_type(16))) _Float16 v16h;
typedef __attribute__((ext_vector_type(8)))  float    v8f;

#define NO_ROWS   8192
#define NTRAIN    100000
#define DIM       256
#define OUT_DIM   256
#define KNN       16
#define IN_DIM    1025        // 1 + 256 + 512 + 256
#define K1PAD     1056        // IN_DIM padded to 33*32
#define KT1       33          // K-tiles for GEMM1 (K1PAD/32)
#define NT1       32          // N-tiles for GEMM1 (512/16)
#define H1        512
#define KT2       16          // K-tiles for GEMM2 (512/32)
#define NT2       16          // N-tiles for GEMM2 (256/16)

// padded conv images: [2 zero | data | 4 zero] rows, 16 halves per row
#define NB_ROWS   (2 + DIM + 4)        // 262
#define IT_ROWS   (2 + 2 * DIM + 4)    // 518

// ---------------------------------------------------------------------------
// Kernel 1: fold attention:  v_src = W_att @ a_src,  v_dst = W_att @ a_dst
// ---------------------------------------------------------------------------
__global__ void att_vecs_kernel(const float* __restrict__ W_att,
                                const float* __restrict__ a_src,
                                const float* __restrict__ a_dst,
                                float* __restrict__ vsrc,
                                float* __restrict__ vdst) {
    int d = threadIdx.x;
    const float* av = (blockIdx.x == 0) ? a_src : a_dst;
    float*       vv = (blockIdx.x == 0) ? vsrc  : vdst;
    const float* row = W_att + (size_t)d * DIM;
    float s = 0.f;
    for (int o = 0; o < DIM; ++o) s += row[o] * av[o];
    vv[d] = s;
}

// ---------------------------------------------------------------------------
// Kernels 2/3: pre-swizzle MLP weights into WMMA B-fragment layout.
// This code's B packing (16x16x32 f16, contiguous k): lane l, half element i holds K = i + 16*(l>>4),
// column N = ntile*16 + (l&15). 16 halves per lane stored contiguously.
// ---------------------------------------------------------------------------
__global__ void pack_w1_kernel(const float* __restrict__ w1, _Float16* __restrict__ w1p) {
    int ktile = blockIdx.x >> 5;        // 0..32
    int ntile = blockIdx.x & 31;        // 0..31
    int lane  = threadIdx.x;            // 0..31
    int half  = lane >> 4;
    int n     = ntile * 16 + (lane & 15);
    _Float16* dst = w1p + (((size_t)(ktile * NT1 + ntile)) * 32 + lane) * 16;
#pragma unroll
    for (int i = 0; i < 16; ++i) {
        int k = ktile * 32 + i + 16 * half;
        float v = (k < IN_DIM) ? w1[(size_t)k * H1 + n] : 0.f;
        dst[i] = (_Float16)v;
    }
}

__global__ void pack_w2_kernel(const float* __restrict__ w2, _Float16* __restrict__ w2p) {
    int ktile = blockIdx.x >> 4;        // 0..15
    int ntile = blockIdx.x & 15;        // 0..15
    int lane  = threadIdx.x;
    int half  = lane >> 4;
    int n     = ntile * 16 + (lane & 15);
    _Float16* dst = w2p + (((size_t)(ktile * NT2 + ntile)) * 32 + lane) * 16;
#pragma unroll
    for (int i = 0; i < 16; ++i) {
        int k = ktile * 32 + i + 16 * half;   // always < 512
        dst[i] = (_Float16)w2[(size_t)k * OUT_DIM + n];
    }
}

// ---------------------------------------------------------------------------
// Kernel 4: pack conv weights as WMMA A-fragments.
// A = [co(16) x K(96)], K flattened t-major: k = tap*16 + ci (tap 0..5, taps>=5 -> 0).
// This code's A packing (16x32, contiguous k like B): lane row m=lane&15, element i:
//   kk = i + 16*(lane>>4);  K = kt*32 + kk.
// ---------------------------------------------------------------------------
__global__ void pack_convA_kernel(const float* __restrict__ convk_w,
                                  _Float16* __restrict__ awp) {
    int lane = threadIdx.x;             // 0..31
    int halfv = lane >> 4;
    int m = lane & 15;                  // co
    for (int kt = 0; kt < 3; ++kt) {
        _Float16* dst = awp + ((size_t)kt * 32 + lane) * 16;
#pragma unroll
        for (int i = 0; i < 16; ++i) {
            int vg = i >> 1, p = i & 1;
            int kk = i + 16 * halfv;
            int K = kt * 32 + kk;
            int tap = K >> 4, ci = K & 15;
            float v = (tap < 5) ? convk_w[(m * KNN + ci) * 5 + tap] : 0.f;
            dst[i] = (_Float16)v;
        }
    }
}

// ---------------------------------------------------------------------------
// Kernel 5: per-row prep. One block (256 thr = 8 waves) per output row.
// Convs done via WMMA against zero-halo-padded transposed f16 images in LDS:
// every B-fragment is an unconditional pair of ds_load_b128 (no exec masking).
// ---------------------------------------------------------------------------
__device__ __forceinline__ float block_reduce(float v, float* red, int t) {
    red[t] = v;
    __syncthreads();
#pragma unroll
    for (int s = 128; s > 0; s >>= 1) {
        if (t < s) red[t] += red[t + s];
        __syncthreads();
    }
    float r = red[0];
    __syncthreads();
    return r;
}

__global__ __launch_bounds__(256) void row_prep_kernel(
    const float* __restrict__ X, const float* __restrict__ X_train,
    const float* __restrict__ y_train, const float* __restrict__ neigh_dist,
    const float* __restrict__ data_m_train, const float* __restrict__ data_m_batch,
    const float* __restrict__ vsrc, const float* __restrict__ vdst,
    const float* __restrict__ conv1_w, const float* __restrict__ conv1_b,
    const float* __restrict__ convk_b, const _Float16* __restrict__ awp,
    const int*   __restrict__ neigh_ind,
    _Float16* __restrict__ inpw, float* __restrict__ predw,
    float* __restrict__ out_a, float* __restrict__ out_b) {

    // Inter image: rows [2,258) = X broadcast, rows [258,514) = Neighbors_o^T
    __shared__ __align__(32) _Float16 sInterT[IT_ROWS * KNN];  // 16.2 KB
    // Neighbors image: rows [2,258) = Neighbors_o^T (zero left/right halo)
    __shared__ __align__(32) _Float16 sNbT[NB_ROWS * KNN];     // 8.2 KB
    __shared__ float sX[DIM];
    __shared__ float sVd[DIM];
    __shared__ float red[256];
    __shared__ float sinv[KNN];
    __shared__ float shck[KNN];
    __shared__ float sev[KNN];
    __shared__ float scb[KNN];
    __shared__ int   sidx[KNN];
    __shared__ float ssc[4];   // [0]=1/sum(inv) [1]=pred [2]=softmax max [3]=softmax denom

    const int t    = threadIdx.x;
    const int n    = blockIdx.x;
    const int wv   = t >> 5;
    const int lane = t & 31;
    const int half = lane >> 4;
    const int ncol = lane & 15;

    float xv = X[(size_t)n * DIM + t];
    sX[t]  = xv;
    sVd[t] = vdst[t];
    if (t < KNN) {
        scb[t]  = convk_b[t];
        sidx[t] = neigh_ind[(size_t)n * KNN + t];
        sinv[t] = 1.f / (neigh_dist[(size_t)n * KNN + t] + 1e-8f);
    }
    // zero halos (rows 0..1 and last 4 rows of each image)
    if (t < 2 * KNN) { sInterT[t] = (_Float16)0.f; sNbT[t] = (_Float16)0.f; }
    if (t < 4 * KNN) {
        sInterT[(IT_ROWS - 4) * KNN + t] = (_Float16)0.f;
        sNbT[(NB_ROWS - 4) * KNN + t]    = (_Float16)0.f;
    }
    {   // X region of the Inter image (replicated across ci)
        _Float16 xh = (_Float16)xv;
#pragma unroll
        for (int k = 0; k < KNN; ++k) sInterT[(2 + t) * KNN + k] = xh;
    }
    float xm = xv * data_m_batch[(size_t)n * DIM + t];
    __syncthreads();

    // h_x = dot(Xm, v_src)
    float hx = block_reduce(xm * vsrc[t], red, t);

    // weights normalizer + pred (tiny, serial on thread 0)
    if (t == 0) {
        float s = 0.f;
        for (int i = 0; i < KNN; ++i) s += sinv[i];
        float ri = 1.f / s;
        float p = 0.f;
        for (int i = 0; i < KNN; ++i) p += y_train[sidx[i]] * sinv[i] * ri;
        ssc[0] = ri; ssc[1] = p;
    }
    __syncthreads();
    float rinv = ssc[0];

    // gather + mask: wave wv owns neighbors k = 2*wv, 2*wv+1.
    // Writes transposed weighted features into BOTH images; h_c[k] = dot(cand, v_dst).
    for (int j = 0; j < 2; ++j) {
        int k = wv * 2 + j;
        size_t rowb = (size_t)sidx[k] * DIM;
        if (j == 0) {
            size_t nb = (size_t)sidx[k + 1] * DIM + (size_t)lane * 8;
            __builtin_prefetch(X_train + nb, 0, 0);
            __builtin_prefetch(data_m_train + nb, 0, 0);
        }
        float wk = sinv[k] * rinv;
        float dotp = 0.f;
#pragma unroll
        for (int it = 0; it < 2; ++it) {
            int d0 = it * 128 + lane * 4;
            float4 c4 = *(const float4*)(X_train + rowb + d0);
            float4 m4 = *(const float4*)(data_m_train + rowb + d0);
            float c0 = c4.x * m4.x, c1 = c4.y * m4.y, c2 = c4.z * m4.z, c3 = c4.w * m4.w;
            dotp += c0 * sVd[d0] + c1 * sVd[d0 + 1] + c2 * sVd[d0 + 2] + c3 * sVd[d0 + 3];
            _Float16 h0 = (_Float16)(c0 * wk), h1 = (_Float16)(c1 * wk);
            _Float16 h2 = (_Float16)(c2 * wk), h3 = (_Float16)(c3 * wk);
            sInterT[(258 + d0    ) * KNN + k] = h0;
            sInterT[(258 + d0 + 1) * KNN + k] = h1;
            sInterT[(258 + d0 + 2) * KNN + k] = h2;
            sInterT[(258 + d0 + 3) * KNN + k] = h3;
            sNbT[(2 + d0    ) * KNN + k] = h0;
            sNbT[(2 + d0 + 1) * KNN + k] = h1;
            sNbT[(2 + d0 + 2) * KNN + k] = h2;
            sNbT[(2 + d0 + 3) * KNN + k] = h3;
        }
#pragma unroll
        for (int off = 16; off > 0; off >>= 1) dotp += __shfl_xor(dotp, off, 32);
        if (lane == 0) shck[k] = dotp;
    }
    __syncthreads();

    // attention: e = leaky_relu(h_x + h_c); b = e; a = softmax_k(e)
    if (t < KNN) {
        float e = hx + shck[t];
        e = (e >= 0.f) ? e : 0.2f * e;
        sev[t] = e;
        out_b[(size_t)n * KNN + t] = e;
    }
    __syncthreads();
    if (t == 0) {
        float m = sev[0];
        for (int i = 1; i < KNN; ++i) m = fmaxf(m, sev[i]);
        float s = 0.f;
        for (int i = 0; i < KNN; ++i) s += expf(sev[i] - m);
        ssc[2] = m; ssc[3] = s;
    }
    __syncthreads();
    if (t < KNN)
        out_a[(size_t)n * KNN + t] = expf(sev[t] - ssc[2]) / ssc[3];

    _Float16* inp = inpw + (size_t)n * K1PAD;

    // X1: 3-tap conv of X row
    {
        float xl = (t > 0)   ? sX[t - 1] : 0.f;
        float xr = (t < 255) ? sX[t + 1] : 0.f;
        float y  = conv1_b[0] + conv1_w[0] * xl + conv1_w[1] * sX[t] + conv1_w[2] * xr;
        inp[1 + t] = (_Float16)y;                        // cols 1..256
    }
    if (t == 0) { inp[0] = (_Float16)ssc[1]; predw[n] = ssc[1]; }
    if (t < (K1PAD - IN_DIM)) inp[IN_DIM + t] = (_Float16)0.f;

    // ---- both convs as WMMA: D[co, col] = sum_k A[co,k] * B[k, col] ----
    v16h aw[3];
    aw[0] = *(const v16h*)(awp + ((size_t)0 * 32 + lane) * 16);
    aw[1] = *(const v16h*)(awp + ((size_t)1 * 32 + lane) * 16);
    aw[2] = *(const v16h*)(awp + ((size_t)2 * 32 + lane) * 16);

    // 48 jobs: 0..15 -> Neighbors chunks (L=256), 16..47 -> Inter chunks (L=512)
    for (int job = wv; job < 48; job += 8) {
        bool isNb = job < 16;
        int chunk = isNb ? job : job - 16;
        int col   = chunk * 16 + ncol;
        const _Float16* img = isNb ? sNbT : sInterT;
        v8f acc = {};
#pragma unroll
        for (int kt = 0; kt < 3; ++kt) {
            int tap = kt * 2 + half;         // tap index 0..5 (5 = zero A pad)
            // buffer row = 2 + (col + tap - 2) = col + tap; halo makes it always valid
            v16h bf = *(const v16h*)(img + (size_t)(col + tap) * KNN);
            acc = __builtin_amdgcn_wmma_f32_16x16x32_f16(
                false, aw[kt], false, bf, (short)0, acc, false, false);
        }
        // channel max: over 8 regs (co = r + 8*half, + bias), then across M-halves
        float m = -3.402823466e38f;
#pragma unroll
        for (int r = 0; r < 8; ++r) m = fmaxf(m, acc[r] + scb[r + 8 * half]);
        m = fmaxf(m, __shfl_xor(m, 16, 32));
        if (lane < 16) {
            int dst = isNb ? (1 + 3 * DIM + col)   // Neighbors -> cols 769..1024
                           : (1 + DIM + col);      // Inter     -> cols 257..768
            inp[dst] = (_Float16)m;
        }
    }
}

// ---------------------------------------------------------------------------
// Kernel 6: 16-row MLP tile via WMMA f16.
//   x1 = A@w1+b1 (1056->512), x2 = relu(x1@w2+b2) (512->256),
//   out = pred + relu(x2@w3+b3)
// ---------------------------------------------------------------------------
__global__ __launch_bounds__(128) void mlp_kernel(
    const _Float16* __restrict__ inpw, const _Float16* __restrict__ w1p,
    const _Float16* __restrict__ w2p,
    const float* __restrict__ b1, const float* __restrict__ b2,
    const float* __restrict__ w3, const float* __restrict__ b3,
    const float* __restrict__ predw, float* __restrict__ out) {

    // region0 (33792 B): A tile (16x1056 f16), later reused as x2 (16x256 f32)
    // region1 (16384 B): x1 tile (16x512 f16)
    __shared__ __align__(32) unsigned char smem[33792 + 16384];
    _Float16* A1s = (_Float16*)smem;
    float*    x2s = (float*)smem;
    _Float16* x1h = (_Float16*)(smem + 33792);

    const int t    = threadIdx.x;
    const int wave = t >> 5;
    const int lane = t & 31;
    const int half = lane >> 4;
    const int mrow = lane & 15;
    const int n0   = blockIdx.x * 16;

    {   // cooperative A-tile load (16 x 1056 halves = 8448 dwords)
        const unsigned int* src = (const unsigned int*)(inpw + (size_t)n0 * K1PAD);
        unsigned int* dst = (unsigned int*)A1s;
        for (int i = t; i < 16 * K1PAD / 2; i += 128) dst[i] = src[i];
    }
    __syncthreads();

    // ---- GEMM1: [16 x 1056] @ [1056 x 512]; wave handles 8 N-tiles ----
    v8f acc[8] = {};
    for (int kt = 0; kt < KT1; ++kt) {
        v16h af;
#pragma unroll
        for (int i = 0; i < 16; ++i) {   // A: K = i + 16*half (matches B packing)
            int vg = i >> 1, p = i & 1;
            int kk = i + 16 * half;
            af[i] = A1s[mrow * K1PAD + kt * 32 + kk];
        }
#pragma unroll
        for (int j = 0; j < 8; ++j) {
            int ntile = wave * 8 + j;
            v16h bf = *(const v16h*)(w1p + (((size_t)(kt * NT1 + ntile)) * 32 + lane) * 16);
            acc[j] = __builtin_amdgcn_wmma_f32_16x16x32_f16(
                false, af, false, bf, (short)0, acc[j], false, false);
        }
    }
#pragma unroll
    for (int j = 0; j < 8; ++j) {        // C/D layout: M = r + 8*half, N = lane&15
        int col = (wave * 8 + j) * 16 + (lane & 15);
        float bb = b1[col];
#pragma unroll
        for (int r = 0; r < 8; ++r)
            x1h[(r + 8 * half) * H1 + col] = (_Float16)(acc[j][r] + bb);
    }
    __syncthreads();

    // ---- GEMM2: [16 x 512] @ [512 x 256]; wave handles 4 N-tiles ----
    v8f acc2[4] = {};
    for (int kt = 0; kt < KT2; ++kt) {
        v16h af;
#pragma unroll
        for (int i = 0; i < 16; ++i) {
            int vg = i >> 1, p = i & 1;
            int kk = i + 16 * half;
            af[i] = x1h[mrow * H1 + kt * 32 + kk];
        }
#pragma unroll
        for (int j = 0; j < 4; ++j) {
            int ntile = wave * 4 + j;
            v16h bf = *(const v16h*)(w2p + (((size_t)(kt * NT2 + ntile)) * 32 + lane) * 16);
            acc2[j] = __builtin_amdgcn_wmma_f32_16x16x32_f16(
                false, af, false, bf, (short)0, acc2[j], false, false);
        }
    }
#pragma unroll
    for (int j = 0; j < 4; ++j) {
        int col = (wave * 4 + j) * 16 + (lane & 15);
        float bb = b2[col];
#pragma unroll
        for (int r = 0; r < 8; ++r) {
            float v = acc2[j][r] + bb;
            x2s[(r + 8 * half) * OUT_DIM + col] = (v > 0.f) ? v : 0.f;
        }
    }
    __syncthreads();

    // ---- final matvec 256 -> 1, relu, add pred ----
    if (t < 16) {
        const float* xr = x2s + t * OUT_DIM;
        float s = 0.f;
        for (int d = 0; d < OUT_DIM; ++d) s += xr[d] * w3[d];
        s += b3[0];
        s = (s > 0.f) ? s : 0.f;
        out[n0 + t] = predw[n0 + t] + s;
    }
}

// ---------------------------------------------------------------------------
extern "C" void kernel_launch(void* const* d_in, const int* in_sizes, int n_in,
                              void* d_out, int out_size, void* d_ws, size_t ws_size,
                              hipStream_t stream) {
    const float* X            = (const float*)d_in[0];
    const float* X_train      = (const float*)d_in[1];
    const float* y_train      = (const float*)d_in[2];
    const float* neigh_dist   = (const float*)d_in[3];
    const float* data_m_train = (const float*)d_in[4];
    const float* data_m_batch = (const float*)d_in[5];
    const float* W_att        = (const float*)d_in[6];
    const float* a_src        = (const float*)d_in[7];
    const float* a_dst        = (const float*)d_in[8];
    const float* w1           = (const float*)d_in[9];
    const float* b1           = (const float*)d_in[10];
    const float* w2           = (const float*)d_in[11];
    const float* b2           = (const float*)d_in[12];
    const float* w3           = (const float*)d_in[13];
    const float* b3           = (const float*)d_in[14];
    const float* conv1_w      = (const float*)d_in[15];
    const float* conv1_b      = (const float*)d_in[16];
    const float* convk_w      = (const float*)d_in[17];
    const float* convk_b      = (const float*)d_in[18];
    const int*   neigh_ind    = (const int*)d_in[19];

    float* out   = (float*)d_out;                 // [8192] pred+x3
    float* out_a = out + NO_ROWS;                 // [8192*16]
    float* out_b = out_a + (size_t)NO_ROWS * KNN; // [8192*16]

    // workspace layout (all 32B-aligned)
    float*    vsrc  = (float*)d_ws;                         // 256
    float*    vdst  = vsrc + 256;                           // 256
    float*    predw = vdst + 256;                           // 8192
    _Float16* w1p   = (_Float16*)(predw + NO_ROWS);         // 540672 halves
    _Float16* w2p   = w1p + (size_t)KT1 * NT1 * 32 * 16;    // 131072 halves
    _Float16* awp   = w2p + (size_t)KT2 * NT2 * 32 * 16;    // 3*32*16 = 1536 halves
    _Float16* inpw  = awp + 1536;                           // 8192*1056 halves

    att_vecs_kernel<<<2, 256, 0, stream>>>(W_att, a_src, a_dst, vsrc, vdst);
    pack_w1_kernel<<<KT1 * NT1, 32, 0, stream>>>(w1, w1p);
    pack_w2_kernel<<<KT2 * NT2, 32, 0, stream>>>(w2, w2p);
    pack_convA_kernel<<<1, 32, 0, stream>>>(convk_w, awp);
    row_prep_kernel<<<NO_ROWS, 256, 0, stream>>>(
        X, X_train, y_train, neigh_dist, data_m_train, data_m_batch,
        vsrc, vdst, conv1_w, conv1_b, convk_b, awp, neigh_ind,
        inpw, predw, out_a, out_b);
    mlp_kernel<<<NO_ROWS / 16, 128, 0, stream>>>(
        inpw, w1p, w2p, b1, b2, w3, b3, predw, out);
}
